// Block_20298015441599
// MI455X (gfx1250) — matrix operand order repaired and checked by reading, not compiled, not run on hardware
//
#include <hip/hip_runtime.h>
#include <hip/hip_bf16.h>
#include <math.h>

#define DMODEL 768
#define DFFN   3072
#define NHEAD  12
#define HDIM   64
#define BATCH  2
#define SEQ    2048
#define MTOK   (BATCH*SEQ)

typedef __attribute__((ext_vector_type(16))) _Float16 v16h;
typedef __attribute__((ext_vector_type(8)))  float    v8f;
typedef __attribute__((ext_vector_type(4)))  float    v4f;

static __device__ __forceinline__ v8f wmma16(v16h a, v16h b, v8f c) {
  // D = A(16x32 f16) * B(32x16 f16) + C(16x16 f32)
  return __builtin_amdgcn_wmma_f32_16x16x32_f16(false, a, false, b, (short)0, c, false, false);
}
static __device__ __forceinline__ v16h ld2x16(const _Float16* p0, const _Float16* p1) {
  union { v16h v; v4f f[2]; } u;
  u.f[0] = *(const v4f*)p0;
  u.f[1] = *(const v4f*)p1;
  return u.v;
}
static __device__ __forceinline__ v8f vzero8() {
  v8f z;
#pragma unroll
  for (int i = 0; i < 8; i++) z[i] = 0.f;
  return z;
}
// CDNA5 async memory->LDS (ASYNCcnt-tracked, no VGPR round trip).
static __device__ __forceinline__ void async_ld_b128(const _Float16* gptr, _Float16* lptr) {
  unsigned lds = (unsigned)(uintptr_t)lptr;   // generic low 32 bits == LDS byte address
  asm volatile("global_load_async_to_lds_b128 %0, %1, off"
               :: "v"(lds), "v"(gptr) : "memory");
}
static __device__ __forceinline__ void wait_async0() {
  asm volatile("s_wait_asynccnt 0x0" ::: "memory");
}

// ---------------------------------------------------------------------------
// Generic GEMM: C[M,N] = A[M,K](f16) @ W[N,K]^T(f16) + bias
// Double-buffered LDS pipeline fed by global_load_async_to_lds_b128.
// mode 0: store f32 to Cf
// mode 1: GELU(exact) then store f16 to Ch
// mode 2: store f16 to Ch with [B,H,S,HD] head relayout (QKV projections)
// ---------------------------------------------------------------------------
__global__ __launch_bounds__(256) void gemm_xwT(const _Float16* __restrict__ A,
                                                const _Float16* __restrict__ W,
                                                const float* __restrict__ bias,
                                                float* __restrict__ Cf,
                                                _Float16* __restrict__ Ch,
                                                int N, int K, int mode) {
  __shared__ __align__(16) _Float16 As[2][128][40];
  __shared__ __align__(16) _Float16 Bs[2][128][40];
  const int tid  = threadIdx.x;
  const int lane = tid & 31, wid = tid >> 5;
  const int wm = wid & 1, wn = wid >> 1;            // 2 x 4 wave grid
  const int m0 = blockIdx.y * 128, n0 = blockIdx.x * 128;
  const int h8  = lane >> 4;                        // 0/1 lane half
  const int l16 = lane & 15;

  v8f acc[4][2];
#pragma unroll
  for (int i = 0; i < 4; i++)
#pragma unroll
    for (int j = 0; j < 2; j++) acc[i][j] = vzero8();

  const int lrow = tid >> 1;            // 0..127
  const int lcol = (tid & 1) * 16;      // 0/16

  auto issue_tile = [&](int kk, int buf) {
    const _Float16* ga = A + (size_t)(m0 + lrow) * K + kk + lcol;
    const _Float16* gb = W + (size_t)(n0 + lrow) * K + kk + lcol;
    async_ld_b128(ga,     &As[buf][lrow][lcol]);
    async_ld_b128(ga + 8, &As[buf][lrow][lcol + 8]);
    async_ld_b128(gb,     &Bs[buf][lrow][lcol]);
    async_ld_b128(gb + 8, &Bs[buf][lrow][lcol + 8]);
  };

  const int nk = K >> 5;
  issue_tile(0, 0);

  for (int kt = 0; kt < nk; kt++) {
    const int cur = kt & 1;
    wait_async0();                 // my async writes into buf[cur] landed
    __syncthreads();               // everyone's landed; prior reads of buf[cur^1] done
    if (kt + 1 < nk) issue_tile((kt + 1) << 5, cur ^ 1);

    // hoist all fragment loads so the 8 WMMAs can issue back-to-back
    v16h afr[4], bfr[2];
#pragma unroll
    for (int ni = 0; ni < 2; ni++) {
      const int n = wn * 32 + ni * 16 + l16;        // B: col=lane, 16 K-contig @ 16*h8
      bfr[ni] = ld2x16(&Bs[cur][n][16 * h8], &Bs[cur][n][16 * h8 + 8]);
    }
#pragma unroll
    for (int mi = 0; mi < 4; mi++) {
      const int m = wm * 64 + mi * 16 + l16;        // A: row=lane, 16 K-contig @ 16*h8
      afr[mi] = ld2x16(&As[cur][m][16 * h8], &As[cur][m][16 * h8 + 8]);
    }
#pragma unroll
    for (int mi = 0; mi < 4; mi++)
#pragma unroll
      for (int ni = 0; ni < 2; ni++)
        acc[mi][ni] = wmma16(afr[mi], bfr[ni], acc[mi][ni]);
  }

#pragma unroll
  for (int ni = 0; ni < 2; ni++) {
    const int cg = n0 + wn * 32 + ni * 16 + l16;
    const float bv = bias[cg];
#pragma unroll
    for (int mi = 0; mi < 4; mi++) {
#pragma unroll
      for (int r = 0; r < 8; r++) {
        const int mg = m0 + wm * 64 + mi * 16 + r + 8 * h8;
        float v = acc[mi][ni][r] + bv;
        if (mode == 0) {
          Cf[(size_t)mg * N + cg] = v;
        } else if (mode == 1) {
          float gl = 0.5f * v * (1.0f + erff(v * 0.70710678118654752f));
          Ch[(size_t)mg * N + cg] = (_Float16)gl;
        } else {
          const int b = mg >> 11, s = mg & (SEQ - 1);
          const int hd_h = cg >> 6, hd = cg & 63;
          Ch[(((size_t)b * NHEAD + hd_h) * SEQ + s) * HDIM + hd] = (_Float16)v;
        }
      }
    }
  }
}

// ---------------------------------------------------------------------------
// Flash attention: 1 WG = (b, h, 64 queries); 4 waves x 16 queries.
// Streams keys in blocks of 32, online softmax, never materializes scores.
// K tile staged via async memory->LDS; V transposed through VGPRs.
// ---------------------------------------------------------------------------
__global__ __launch_bounds__(128) void flash_attn(const _Float16* __restrict__ Q,
                                                  const _Float16* __restrict__ Kg,
                                                  const _Float16* __restrict__ Vg,
                                                  const float* __restrict__ mask,
                                                  _Float16* __restrict__ O) {
  __shared__ __align__(16) _Float16 Ks[32][72];       // [key][hd]
  __shared__ __align__(16) _Float16 Vt[64][40];       // [hd][key] (transposed)
  __shared__ __align__(16) _Float16 Pw[4][16][40];    // per-wave P^T relayout [q][key]
  __shared__ float smask[32];

  const int tid  = threadIdx.x;
  const int lane = tid & 31, wid = tid >> 5;
  const int h8 = lane >> 4, l16 = lane & 15;
  const int b = blockIdx.z, hd_h = blockIdx.y;
  const int q0 = blockIdx.x * 64 + wid * 16;
  const size_t headbase = ((size_t)b * NHEAD + hd_h) * SEQ * HDIM;

  // Q^T as B-operand: two K=32 chunks covering HDIM=64. lane col=q, K(hd) contig.
  v16h qf[2];
  {
    const _Float16* qp = Q + headbase + (size_t)(q0 + l16) * HDIM;
#pragma unroll
    for (int c = 0; c < 2; c++)
      qf[c] = ld2x16(qp + 32 * c + 16 * h8, qp + 32 * c + 16 * h8 + 8);
  }

  v8f oacc[4];
#pragma unroll
  for (int c = 0; c < 4; c++) oacc[c] = vzero8();
  float mrun = -1e30f, lrun = 0.f;

  const int krow = tid >> 2;         // 0..31
  const int kc16 = (tid & 3) * 16;   // 0,16,32,48

  for (int j = 0; j < SEQ; j += 32) {
    __syncthreads();
    {   // stage K tile [32 keys][64 hd] via async LDS DMA
      const _Float16* kp = Kg + headbase + (size_t)(j + krow) * HDIM + kc16;
      async_ld_b128(kp,     &Ks[krow][kc16]);
      async_ld_b128(kp + 8, &Ks[krow][kc16 + 8]);
    }
    {   // stage V tile transposed -> Vt[hd][key]
      const _Float16* vp = Vg + headbase + (size_t)(j + krow) * HDIM + kc16;
      union { v4f f[2]; _Float16 h[16]; } u;
      u.f[0] = *(const v4f*)vp;
      u.f[1] = *(const v4f*)(vp + 8);
#pragma unroll
      for (int i = 0; i < 16; i++) Vt[kc16 + i][krow] = u.h[i];
    }
    if (tid < 32) smask[tid] = -10000.f * (1.f - mask[(size_t)b * SEQ + j + tid]);
    wait_async0();
    __syncthreads();

    // S^T (keys x queries) : two 16-key tiles, K-dim = HDIM via 2 WMMAs each.
    // Hoist all K fragments, then run the 4 WMMAs back-to-back.
    v16h kf[2][2];
#pragma unroll
    for (int t = 0; t < 2; t++) {
      const int key = 16 * t + l16;
#pragma unroll
      for (int c = 0; c < 2; c++)
        kf[t][c] = ld2x16(&Ks[key][32 * c + 16 * h8], &Ks[key][32 * c + 16 * h8 + 8]);
    }
    v8f st[2];
#pragma unroll
    for (int t = 0; t < 2; t++) {
      v8f s = vzero8();
      s = wmma16(kf[t][0], qf[0], s);
      s = wmma16(kf[t][1], qf[1], s);
      st[t] = s;
    }
#pragma unroll
    for (int t = 0; t < 2; t++)
#pragma unroll
      for (int r = 0; r < 8; r++)
        st[t][r] = st[t][r] * 0.125f + smask[16 * t + r + 8 * h8];  // 1/sqrt(64), mask

    // online softmax stats for this lane's query column (partner = lane ^ 16)
    float mloc = -1e30f;
#pragma unroll
    for (int t = 0; t < 2; t++)
#pragma unroll
      for (int r = 0; r < 8; r++) mloc = fmaxf(mloc, st[t][r]);
    mloc = fmaxf(mloc, __shfl_xor(mloc, 16, 32));
    const float mnew  = fmaxf(mrun, mloc);
    const float alpha = __expf(mrun - mnew);
    float lsum = 0.f;
#pragma unroll
    for (int t = 0; t < 2; t++)
#pragma unroll
      for (int r = 0; r < 8; r++) {
        float p = __expf(st[t][r] - mnew);
        st[t][r] = p;
        lsum += p;
      }
    lsum += __shfl_xor(lsum, 16, 32);
    lrun = lrun * alpha + lsum;
    mrun = mnew;

    // store P^T into wave-private LDS as [q][key] (packed f16 pairs)
#pragma unroll
    for (int t = 0; t < 2; t++) {
      const int kb = 16 * t + 8 * h8;
#pragma unroll
      for (int rp = 0; rp < 4; rp++) {
        union { _Float16 h[2]; unsigned u; } pk;
        pk.h[0] = (_Float16)st[t][2 * rp];
        pk.h[1] = (_Float16)st[t][2 * rp + 1];
        *(unsigned*)&Pw[wid][l16][kb + 2 * rp] = pk.u;
      }
    }

    // rescale running O by alpha (per query row)
#pragma unroll
    for (int r = 0; r < 8; r++) {
      const float ar = __shfl(alpha, r + 8 * h8, 32);
#pragma unroll
      for (int c = 0; c < 4; c++) oacc[c][r] *= ar;
    }

    // O += P(16x32) @ V(32x64) : hoist loads, then 4 WMMAs
    {
      v16h pf = ld2x16(&Pw[wid][l16][16 * h8], &Pw[wid][l16][16 * h8 + 8]);
      v16h vf[4];
#pragma unroll
      for (int c = 0; c < 4; c++)
        vf[c] = ld2x16(&Vt[16 * c + l16][16 * h8], &Vt[16 * c + l16][16 * h8 + 8]);
#pragma unroll
      for (int c = 0; c < 4; c++)
        oacc[c] = wmma16(pf, vf[c], oacc[c]);
    }
  }

  const float inv = 1.f / lrun;
#pragma unroll
  for (int r = 0; r < 8; r++) {
    const float ir = __shfl(inv, r + 8 * h8, 32);
    const int tok = q0 + r + 8 * h8;
    _Float16* op = O + ((size_t)(b * SEQ + tok)) * DMODEL + hd_h * HDIM;
#pragma unroll
    for (int c = 0; c < 4; c++)
      op[16 * c + l16] = (_Float16)(oacc[c][r] * ir);
  }
}

// ---------------------------------------------------------------------------
// Fused residual + LayerNorm. One wave per 768-wide row (24 elems/lane).
// ---------------------------------------------------------------------------
__global__ __launch_bounds__(256) void ln_residual(const float* __restrict__ A,
                                                   const float* __restrict__ R,
                                                   const float* __restrict__ g,
                                                   const float* __restrict__ be,
                                                   float* __restrict__ o32,
                                                   _Float16* __restrict__ o16) {
  const int lane = threadIdx.x & 31, wid = threadIdx.x >> 5;
  const int row = blockIdx.x * 8 + wid;
  const size_t base = (size_t)row * DMODEL;
  float v[24];
  float s = 0.f;
#pragma unroll
  for (int i = 0; i < 24; i++) {
    int idx = lane + 32 * i;
    v[i] = A[base + idx] + R[base + idx];
    s += v[i];
  }
  for (int m = 16; m >= 1; m >>= 1) s += __shfl_xor(s, m, 32);
  const float mean = s * (1.f / (float)DMODEL);
  float q = 0.f;
#pragma unroll
  for (int i = 0; i < 24; i++) { float d = v[i] - mean; q += d * d; }
  for (int m = 16; m >= 1; m >>= 1) q += __shfl_xor(q, m, 32);
  const float rstd = rsqrtf(q * (1.f / (float)DMODEL) + 1e-12f);
#pragma unroll
  for (int i = 0; i < 24; i++) {
    int idx = lane + 32 * i;
    float y = g[idx] * ((v[i] - mean) * rstd) + be[idx];
    o32[base + idx] = y;
    if (o16) o16[base + idx] = (_Float16)y;
  }
}

__global__ void cvt_f16(const float* __restrict__ in, _Float16* __restrict__ out, long n) {
  long i = (long)blockIdx.x * blockDim.x + threadIdx.x;
  long stride = (long)gridDim.x * blockDim.x;
  for (; i < n; i += stride) out[i] = (_Float16)in[i];
}

// ---------------------------------------------------------------------------
extern "C" void kernel_launch(void* const* d_in, const int* in_sizes, int n_in,
                              void* d_out, int out_size, void* d_ws, size_t ws_size,
                              hipStream_t stream) {
  (void)in_sizes; (void)n_in; (void)out_size; (void)ws_size;
  const float* x    = (const float*)d_in[0];
  const float* mask = (const float*)d_in[1];
  const float* Wq = (const float*)d_in[2];  const float* bq  = (const float*)d_in[3];
  const float* Wk = (const float*)d_in[4];  const float* bk  = (const float*)d_in[5];
  const float* Wv = (const float*)d_in[6];  const float* bv  = (const float*)d_in[7];
  const float* Wp = (const float*)d_in[8];  const float* bp  = (const float*)d_in[9];
  const float* g1 = (const float*)d_in[10]; const float* be1 = (const float*)d_in[11];
  const float* W1 = (const float*)d_in[12]; const float* bf1 = (const float*)d_in[13];
  const float* W2 = (const float*)d_in[14]; const float* bf2 = (const float*)d_in[15];
  const float* g2 = (const float*)d_in[16]; const float* be2 = (const float*)d_in[17];
  float* out = (float*)d_out;

  char* ws = (char*)d_ws;
  size_t off = 0;
  auto take = [&](size_t bytes) -> char* {
    char* p = ws + off;
    off += (bytes + 255) & ~(size_t)255;
    return p;
  };
  _Float16* xh   = (_Float16*)take((size_t)MTOK * DMODEL * 2);
  _Float16* wqh  = (_Float16*)take((size_t)DMODEL * DMODEL * 2);
  _Float16* wkh  = (_Float16*)take((size_t)DMODEL * DMODEL * 2);
  _Float16* wvh  = (_Float16*)take((size_t)DMODEL * DMODEL * 2);
  _Float16* wph  = (_Float16*)take((size_t)DMODEL * DMODEL * 2);
  _Float16* w1h  = (_Float16*)take((size_t)DFFN * DMODEL * 2);
  _Float16* w2h  = (_Float16*)take((size_t)DMODEL * DFFN * 2);
  _Float16* qh   = (_Float16*)take((size_t)MTOK * DMODEL * 2);   // [B,H,S,HD]
  _Float16* kh   = (_Float16*)take((size_t)MTOK * DMODEL * 2);
  _Float16* vh   = (_Float16*)take((size_t)MTOK * DMODEL * 2);
  _Float16* hcat = (_Float16*)take((size_t)MTOK * DMODEL * 2);   // [B,S,D]
  float*    tmp  = (float*)   take((size_t)MTOK * DMODEL * 4);   // proj / ffn2 out
  float*    x1   = (float*)   take((size_t)MTOK * DMODEL * 4);
  _Float16* x1h  = (_Float16*)take((size_t)MTOK * DMODEL * 2);
  _Float16* ffh  = (_Float16*)take((size_t)MTOK * DFFN * 2);

  // f32 -> f16 staging
  cvt_f16<<<2048, 256, 0, stream>>>(x,  xh,  (long)MTOK * DMODEL);
  cvt_f16<<<512,  256, 0, stream>>>(Wq, wqh, (long)DMODEL * DMODEL);
  cvt_f16<<<512,  256, 0, stream>>>(Wk, wkh, (long)DMODEL * DMODEL);
  cvt_f16<<<512,  256, 0, stream>>>(Wv, wvh, (long)DMODEL * DMODEL);
  cvt_f16<<<512,  256, 0, stream>>>(Wp, wph, (long)DMODEL * DMODEL);
  cvt_f16<<<2048, 256, 0, stream>>>(W1, w1h, (long)DFFN * DMODEL);
  cvt_f16<<<2048, 256, 0, stream>>>(W2, w2h, (long)DMODEL * DFFN);

  dim3 gq(DMODEL / 128, MTOK / 128);       // (6, 32)
  gemm_xwT<<<gq, 256, 0, stream>>>(xh, wqh, bq, nullptr, qh, DMODEL, DMODEL, 2);
  gemm_xwT<<<gq, 256, 0, stream>>>(xh, wkh, bk, nullptr, kh, DMODEL, DMODEL, 2);
  gemm_xwT<<<gq, 256, 0, stream>>>(xh, wvh, bv, nullptr, vh, DMODEL, DMODEL, 2);

  dim3 ga(SEQ / 64, NHEAD, BATCH);         // (32, 12, 2)
  flash_attn<<<ga, 128, 0, stream>>>(qh, kh, vh, mask, hcat);

  gemm_xwT<<<gq, 256, 0, stream>>>(hcat, wph, bp, tmp, nullptr, DMODEL, DMODEL, 0);
  ln_residual<<<MTOK / 8, 256, 0, stream>>>(tmp, x, g1, be1, x1, x1h);

  dim3 gf1(DFFN / 128, MTOK / 128);        // (24, 32)
  gemm_xwT<<<gf1, 256, 0, stream>>>(x1h, w1h, bf1, nullptr, ffh, DFFN, DMODEL, 1);
  gemm_xwT<<<gq, 256, 0, stream>>>(ffh, w2h, bf2, tmp, nullptr, DMODEL, DFFN, 0);
  ln_residual<<<MTOK / 8, 256, 0, stream>>>(tmp, x1, g2, be2, out, nullptr);
}
